// GraphNetBlock_77318001263114
// MI455X (gfx1250) — compile-verified, matrix operand order checked by reading, not run on hardware
//
#include <hip/hip_runtime.h>

#define N_NODES 50000
#define N_EDGES 400000
#define D 128

typedef __attribute__((ext_vector_type(16))) __bf16 v16bf;
typedef __attribute__((ext_vector_type(8)))  float  v8f;

union FragBF { uint4 u[2]; v16bf v; };

__device__ __forceinline__ unsigned short f2bf(float f) {
  unsigned int u = __float_as_uint(f);
  u += 0x7FFFu + ((u >> 16) & 1u);           // round-to-nearest-even
  return (unsigned short)(u >> 16);
}

__device__ __forceinline__ void zero4(v8f acc[4]) {
  const v8f z = {0.f, 0.f, 0.f, 0.f, 0.f, 0.f, 0.f, 0.f};
#pragma unroll
  for (int i = 0; i < 4; ++i) acc[i] = z;
}

// One wave computes a 16-row x 64-col patch: acc[nt] for nt=0..3 (16 cols each).
// A tile in LDS, bf16, row stride = KDIM. Weights pre-packed in B-fragment order.
template<int KDIM>
__device__ __forceinline__ void gemm4(const unsigned short* __restrict__ Alds,
                                      int rbase, int lane,
                                      const unsigned int* __restrict__ Wp,
                                      int ntBase, v8f acc[4]) {
  const int nkc = KDIM / 32;
  const int m = rbase + (lane & 15);
  const int h = lane >> 4;
  for (int kc = 0; kc < nkc; ++kc) {
    FragBF fa;
    fa.u[0] = *(const uint4*)(Alds + m * KDIM + kc * 32 + h * 8);        // K = h*8 .. +7
    fa.u[1] = *(const uint4*)(Alds + m * KDIM + kc * 32 + 16 + h * 8);   // K = 16+h*8 .. +7
#pragma unroll
    for (int nt = 0; nt < 4; ++nt) {
      FragBF fb;
      const uint4* b = (const uint4*)(Wp + (((size_t)(ntBase + nt) * nkc + kc) * 32 + lane) * 8);
      fb.u[0] = b[0]; fb.u[1] = b[1];
      acc[nt] = __builtin_amdgcn_wmma_f32_16x16x32_bf16(
          false, fa.v, false, fb.v, (short)0, acc[nt], false, false);
    }
  }
}

__device__ __forceinline__ void relu_store_bf16(v8f acc[4], const float* __restrict__ bias,
                                                int colBase, int rbase, int lane,
                                                unsigned short* __restrict__ Hout) {
  const int n = lane & 15, h = lane >> 4;
#pragma unroll
  for (int nt = 0; nt < 4; ++nt) {
    const int col = colBase + nt * 16 + n;
    const float bv = bias[col];
#pragma unroll
    for (int r = 0; r < 8; ++r) {
      float v = acc[nt][r] + bv;
      v = v > 0.f ? v : 0.f;
      Hout[(rbase + r + 8 * h) * D + col] = f2bf(v);
    }
  }
}

__device__ __forceinline__ void f32_store(v8f acc[4], const float* __restrict__ bias,
                                          int colBase, int rbase, int lane,
                                          float* __restrict__ Fout) {
  const int n = lane & 15, h = lane >> 4;
#pragma unroll
  for (int nt = 0; nt < 4; ++nt) {
    const int col = colBase + nt * 16 + n;
    const float bv = bias[col];
#pragma unroll
    for (int r = 0; r < 8; ++r)
      Fout[(rbase + r + 8 * h) * D + col] = acc[nt][r] + bv;
  }
}

// ---- weight pre-pack: fp32 (K x 128 row-major) -> bf16 WMMA B-fragment order ----
__global__ __launch_bounds__(256) void pack_w_kernel(const float* __restrict__ W,
                                                     unsigned int* __restrict__ out,
                                                     int K, int total) {
  int p = blockIdx.x * 256 + threadIdx.x;
  if (p >= total) return;
  const int nkc  = K >> 5;
  const int d    = p & 7;
  const int lane = (p >> 3) & 31;
  const int rest = p >> 8;
  const int kc   = rest % nkc;
  const int nt   = rest / nkc;
  const int h    = lane >> 4;
  const int n    = (nt << 4) + (lane & 15);
  const int base = (d < 4) ? (h * 8 + 2 * d) : (16 + h * 8 + 2 * (d - 4));
  const int k0   = kc * 32 + base;
  unsigned int lo = f2bf(W[(size_t)k0 * D + n]);
  unsigned int hi = f2bf(W[(size_t)(k0 + 1) * D + n]);
  out[p] = lo | (hi << 16);
}

// ---- edge MLP: gather + 3-layer MLP + LayerNorm + atomic segment-sum + residual ----
__global__ __launch_bounds__(256) void edge_mlp_kernel(
    const float* __restrict__ nodef, const float* __restrict__ edgef,
    const int* __restrict__ senders, const int* __restrict__ receivers,
    const unsigned int* __restrict__ W1p, const unsigned int* __restrict__ W2p,
    const unsigned int* __restrict__ W3p,
    const float* __restrict__ b1, const float* __restrict__ b2, const float* __restrict__ b3,
    const float* __restrict__ gam, const float* __restrict__ bet,
    float* __restrict__ agg, float* __restrict__ outE) {
  __shared__ __align__(16) unsigned char smem[65536];
  unsigned short* A0 = (unsigned short*)smem;            // [64][384] bf16, 48KB
  unsigned short* H1 = (unsigned short*)(smem + 49152);  // [64][128] bf16, 16KB
  unsigned short* H2 = (unsigned short*)smem;            // alias A0[0:16KB] (A0 dead after L1)
  float*          F  = (float*)(smem + 16384);           // [64][128] f32 (alias A0[16K:48K])

  const int tid  = threadIdx.x;
  const int lane = tid & 31;
  const int wave = tid >> 5;
  const int rbase   = (wave >> 1) * 16;
  const int colBase = (wave & 1) * 64;
  const int ntBase  = (wave & 1) * 4;
  const size_t eBase = (size_t)blockIdx.x * 64;

  // gather [node[s] | node[r] | edge] -> bf16 tile (2 cols per dword)
  unsigned int* A0u = (unsigned int*)A0;
  for (int idx = tid; idx < 64 * 192; idx += 256) {
    const int row = idx / 192;
    const int c   = (idx % 192) * 2;
    const size_t e = eBase + row;
    const float* src;
    if (c < 128)      src = nodef + (size_t)senders[e] * D + c;
    else if (c < 256) src = nodef + (size_t)receivers[e] * D + (c - 128);
    else              src = edgef + e * D + (c - 256);
    unsigned int lo = f2bf(src[0]);
    unsigned int hi = f2bf(src[1]);
    A0u[idx] = lo | (hi << 16);
  }
  __syncthreads();

  v8f acc[4];
  zero4(acc);
  gemm4<384>(A0, rbase, lane, W1p, ntBase, acc);
  relu_store_bf16(acc, b1, colBase, rbase, lane, H1);
  __syncthreads();

  zero4(acc);
  gemm4<128>(H1, rbase, lane, W2p, ntBase, acc);
  relu_store_bf16(acc, b2, colBase, rbase, lane, H2);
  __syncthreads();

  zero4(acc);
  gemm4<128>(H2, rbase, lane, W3p, ntBase, acc);
  f32_store(acc, b3, colBase, rbase, lane, F);
  __syncthreads();

  // LayerNorm: 4 threads per row, butterfly reduce over 128 cols
  {
    const int row = tid >> 2;
    const int q   = tid & 3;
    const size_t e = eBase + row;
    const float* Fr = F + row * D + q * 32;
    float s = 0.f, ss = 0.f;
#pragma unroll
    for (int c = 0; c < 32; ++c) { float x = Fr[c]; s += x; ss += x * x; }
    s += __shfl_xor(s, 1); ss += __shfl_xor(ss, 1);
    s += __shfl_xor(s, 2); ss += __shfl_xor(ss, 2);
    const float mean = s * (1.f / 128.f);
    const float var  = ss * (1.f / 128.f) - mean * mean;
    const float rstd = rsqrtf(var + 1e-6f);
    const int   rcv  = receivers[e];
    const size_t eo  = e * D;
#pragma unroll
    for (int c0 = 0; c0 < 32; ++c0) {
      const int c = q * 32 + c0;
      const float y = (Fr[c0] - mean) * rstd * gam[c] + bet[c];
      __hip_atomic_fetch_add(&agg[(size_t)rcv * D + c], y,
                             __ATOMIC_RELAXED, __HIP_MEMORY_SCOPE_AGENT);
      outE[eo + c] = y + edgef[eo + c];
    }
  }
}

// ---- node MLP: concat(node, agg) + 3-layer MLP + LayerNorm + residual ----
__global__ __launch_bounds__(256) void node_mlp_kernel(
    const float* __restrict__ nodef, const float* __restrict__ agg,
    const unsigned int* __restrict__ W1p, const unsigned int* __restrict__ W2p,
    const unsigned int* __restrict__ W3p,
    const float* __restrict__ b1, const float* __restrict__ b2, const float* __restrict__ b3,
    const float* __restrict__ gam, const float* __restrict__ bet,
    float* __restrict__ outN) {
  __shared__ __align__(16) unsigned char smem[65536];
  unsigned short* A0 = (unsigned short*)smem;            // [64][256] bf16, 32KB
  unsigned short* H1 = (unsigned short*)(smem + 32768);  // 16KB
  unsigned short* H2 = (unsigned short*)(smem + 49152);  // 16KB
  float*          F  = (float*)smem;                     // alias A0 (dead after L1)

  const int tid  = threadIdx.x;
  const int lane = tid & 31;
  const int wave = tid >> 5;
  const int rbase   = (wave >> 1) * 16;
  const int colBase = (wave & 1) * 64;
  const int ntBase  = (wave & 1) * 4;
  const size_t nBase = (size_t)blockIdx.x * 64;

  unsigned int* A0u = (unsigned int*)A0;
  for (int idx = tid; idx < 64 * 128; idx += 256) {
    const int row = idx >> 7;
    const int c   = (idx & 127) * 2;
    const size_t nd = nBase + row;
    unsigned int v = 0u;
    if (nd < N_NODES) {
      const float* src = (c < 128) ? (nodef + nd * D + c) : (agg + nd * D + (c - 128));
      unsigned int lo = f2bf(src[0]);
      unsigned int hi = f2bf(src[1]);
      v = lo | (hi << 16);
    }
    A0u[idx] = v;
  }
  __syncthreads();

  v8f acc[4];
  zero4(acc);
  gemm4<256>(A0, rbase, lane, W1p, ntBase, acc);
  relu_store_bf16(acc, b1, colBase, rbase, lane, H1);
  __syncthreads();

  zero4(acc);
  gemm4<128>(H1, rbase, lane, W2p, ntBase, acc);
  relu_store_bf16(acc, b2, colBase, rbase, lane, H2);
  __syncthreads();

  zero4(acc);
  gemm4<128>(H2, rbase, lane, W3p, ntBase, acc);
  f32_store(acc, b3, colBase, rbase, lane, F);
  __syncthreads();

  {
    const int row = tid >> 2;
    const int q   = tid & 3;
    const size_t nd = nBase + row;
    const float* Fr = F + row * D + q * 32;
    float s = 0.f, ss = 0.f;
#pragma unroll
    for (int c = 0; c < 32; ++c) { float x = Fr[c]; s += x; ss += x * x; }
    s += __shfl_xor(s, 1); ss += __shfl_xor(ss, 1);
    s += __shfl_xor(s, 2); ss += __shfl_xor(ss, 2);
    const float mean = s * (1.f / 128.f);
    const float var  = ss * (1.f / 128.f) - mean * mean;
    const float rstd = rsqrtf(var + 1e-6f);
    if (nd < N_NODES) {
      const size_t no = nd * D;
#pragma unroll
      for (int c0 = 0; c0 < 32; ++c0) {
        const int c = q * 32 + c0;
        const float y = (Fr[c0] - mean) * rstd * gam[c] + bet[c];
        outN[no + c] = y + nodef[no + c];
      }
    }
  }
}

extern "C" void kernel_launch(void* const* d_in, const int* in_sizes, int n_in,
                              void* d_out, int out_size, void* d_ws, size_t ws_size,
                              hipStream_t stream) {
  (void)in_sizes; (void)n_in; (void)out_size; (void)ws_size;
  const float* nodef = (const float*)d_in[0];
  const float* edgef = (const float*)d_in[1];
  const int*   snd   = (const int*)d_in[2];
  const int*   rcv   = (const int*)d_in[3];
  const float* We1 = (const float*)d_in[4];  const float* be1 = (const float*)d_in[5];
  const float* We2 = (const float*)d_in[6];  const float* be2 = (const float*)d_in[7];
  const float* We3 = (const float*)d_in[8];  const float* be3 = (const float*)d_in[9];
  const float* ge  = (const float*)d_in[10]; const float* Be  = (const float*)d_in[11];
  const float* Wn1 = (const float*)d_in[12]; const float* bn1 = (const float*)d_in[13];
  const float* Wn2 = (const float*)d_in[14]; const float* bn2 = (const float*)d_in[15];
  const float* Wn3 = (const float*)d_in[16]; const float* bn3 = (const float*)d_in[17];
  const float* gn  = (const float*)d_in[18]; const float* Bn  = (const float*)d_in[19];

  char* ws = (char*)d_ws;
  const size_t aggBytes = (size_t)N_NODES * D * sizeof(float);   // 25.6 MB
  float* agg = (float*)ws;
  unsigned int* W1p  = (unsigned int*)(ws + aggBytes);           // 384*64 dwords
  unsigned int* W2p  = W1p + 384 * 64;
  unsigned int* W3p  = W2p + 128 * 64;
  unsigned int* Wn1p = W3p + 128 * 64;                           // 256*64 dwords
  unsigned int* Wn2p = Wn1p + 256 * 64;
  unsigned int* Wn3p = Wn2p + 128 * 64;

  // pack weights into WMMA B-fragment layout (bf16)
  pack_w_kernel<<<(384 * 64 + 255) / 256, 256, 0, stream>>>(We1, W1p, 384, 384 * 64);
  pack_w_kernel<<<(128 * 64 + 255) / 256, 256, 0, stream>>>(We2, W2p, 128, 128 * 64);
  pack_w_kernel<<<(128 * 64 + 255) / 256, 256, 0, stream>>>(We3, W3p, 128, 128 * 64);
  pack_w_kernel<<<(256 * 64 + 255) / 256, 256, 0, stream>>>(Wn1, Wn1p, 256, 256 * 64);
  pack_w_kernel<<<(128 * 64 + 255) / 256, 256, 0, stream>>>(Wn2, Wn2p, 128, 128 * 64);
  pack_w_kernel<<<(128 * 64 + 255) / 256, 256, 0, stream>>>(Wn3, Wn3p, 128, 128 * 64);

  hipMemsetAsync(agg, 0, aggBytes, stream);

  float* outN = (float*)d_out;
  float* outE = outN + (size_t)N_NODES * D;

  edge_mlp_kernel<<<N_EDGES / 64, 256, 0, stream>>>(
      nodef, edgef, snd, rcv, W1p, W2p, W3p, be1, be2, be3, ge, Be, agg, outE);
  node_mlp_kernel<<<(N_NODES + 63) / 64, 256, 0, stream>>>(
      nodef, agg, Wn1p, Wn2p, Wn3p, bn1, bn2, bn3, gn, Bn, outN);
}
